// Attention_59665685676260
// MI455X (gfx1250) — matrix operand order repaired and checked by reading, not compiled, not run on hardware
//
#include <hip/hip_runtime.h>
#include <hip/hip_bf16.h>

// ---------------------------------------------------------------------------
// Single-head causal attention for MI455X (gfx1250, wave32, WMMA).
//   B=4, T=4096, E=1024, H=64.
//   Kernel 0: weight prep  W[E,H] f32 -> Wt[H,E] bf16 (projection B-frags
//             become single contiguous 32B loads).
//   Kernel 1: QKV projection, bf16 WMMA / f32 accumulate, software-pipelined
//             (next A-frag and next B-frags preloaded ahead of the WMMAs).
//             q pre-scaled by 1/sqrt(H); v stored transposed [B,H,T].
//   Kernel 2: flash attention, one wave per 16 query rows, 32-key chunks,
//             online softmax (row max via xor-shuffles, row sum via a
//             WMMA against an all-ones B-fragment), P staged through
//             wave-private LDS to convert C-layout -> A-layout for P@V.
// ---------------------------------------------------------------------------

typedef __bf16 bf16_t;
typedef __attribute__((ext_vector_type(16))) __bf16 v16bf;
typedef __attribute__((ext_vector_type(8)))  __bf16 v8bf;
typedef __attribute__((ext_vector_type(8)))  float  v8f;

#define B_   4
#define T_   4096
#define E_   1024
#define H_   64
#define NROW (B_ * T_)   // 16384 total rows

static __device__ inline v8f vzero8() {
  v8f z = {0.f, 0.f, 0.f, 0.f, 0.f, 0.f, 0.f, 0.f};
  return z;
}

#define WMMA_BF16(A, B, C) \
  __builtin_amdgcn_wmma_f32_16x16x32_bf16(false, (A), false, (B), (short)0, (C), false, false)

// A-fragment (16x32, bf16) from row-major bf16 memory (global or LDS).
// Lane l: row m = l&15.  halves j=0..15 -> K = 16*hi + j (contiguous).
static __device__ inline v16bf load_a_bf16(const bf16_t* base, int stride, int lane) {
  const int m  = lane & 15;
  const int hi = lane >> 4;
  const bf16_t* p = base + m * stride;
  v8bf lo = *(const v8bf*)(p + 16 * hi);
  v8bf hh = *(const v8bf*)(p + 16 * hi + 8);
  union { v16bf v; v8bf h[2]; } u;
  u.h[0] = lo;
  u.h[1] = hh;
  return u.v;
}

// A-fragment from row-major f32 x (convert to bf16). Row stride = E_.
static __device__ inline v16bf load_x_afrag(const float* xrows, int e0, int lane) {
  const int m  = lane & 15;
  const int hi = lane >> 4;
  const float* p = xrows + m * E_ + e0;
  const float4 a0 = *(const float4*)(p + 16 * hi);
  const float4 a1 = *(const float4*)(p + 16 * hi + 4);
  const float4 b0 = *(const float4*)(p + 16 * hi + 8);
  const float4 b1 = *(const float4*)(p + 16 * hi + 12);
  v16bf a;
  a[0]  = (bf16_t)a0.x; a[1]  = (bf16_t)a0.y; a[2]  = (bf16_t)a0.z; a[3]  = (bf16_t)a0.w;
  a[4]  = (bf16_t)a1.x; a[5]  = (bf16_t)a1.y; a[6]  = (bf16_t)a1.z; a[7]  = (bf16_t)a1.w;
  a[8]  = (bf16_t)b0.x; a[9]  = (bf16_t)b0.y; a[10] = (bf16_t)b0.z; a[11] = (bf16_t)b0.w;
  a[12] = (bf16_t)b1.x; a[13] = (bf16_t)b1.y; a[14] = (bf16_t)b1.z; a[15] = (bf16_t)b1.w;
  return a;
}

// B-fragment (32x16, bf16) from transposed bf16 weight Wt [H_, E_].
// Lane l: col h = 16*t + (l&15); half j: row e = e0 + 16*hi + j (contiguous).
static __device__ inline v16bf load_wt_bfrag(const bf16_t* Wt, int e0, int t, int lane) {
  const int n  = lane & 15;
  const int hi = lane >> 4;
  return *(const v16bf*)(Wt + (size_t)(16 * t + n) * E_ + e0 + 16 * hi);
}

// ---------------------------------------------------------------------------
// Kernel 0: convert+transpose weights. grid = (E_*H_/256, 3), block = 256.
// ---------------------------------------------------------------------------
__global__ __launch_bounds__(256) void wprep_kernel(
    const float* __restrict__ Wq, const float* __restrict__ Wk,
    const float* __restrict__ Wv, bf16_t* __restrict__ Wt) {
  const int idx = blockIdx.x * 256 + threadIdx.x;  // 0 .. E_*H_-1
  const int e = idx / H_;
  const int h = idx - e * H_;
  const float* W = (blockIdx.y == 0) ? Wq : (blockIdx.y == 1) ? Wk : Wv;
  bf16_t* dst = Wt + (size_t)blockIdx.y * E_ * H_;
  dst[(size_t)h * E_ + e] = (bf16_t)W[idx];  // coalesced read, strided write
}

// ---------------------------------------------------------------------------
// Kernel 1: QKV projection. 4 waves/block, each wave does one 16-row tile.
// Software-pipelined: A-frag for chunk e0+32 and B-frags for tile t+1 are
// issued before the WMMAs that consume the current ones.
// ---------------------------------------------------------------------------
__global__ __launch_bounds__(128) void qkv_proj_kernel(
    const float* __restrict__ x, const bf16_t* __restrict__ Wt,
    bf16_t* __restrict__ qh, bf16_t* __restrict__ kh, bf16_t* __restrict__ vth) {
  const int lane = threadIdx.x & 31;
  const int wv   = threadIdx.x >> 5;
  const int tile = blockIdx.x * 4 + wv;  // 0..1023
  const int ib   = tile * 16;            // base row in [0, NROW)
  const int n    = lane & 15;
  const int hi   = lane >> 4;

  const bf16_t* Wtq = Wt;
  const bf16_t* Wtk = Wt + (size_t)E_ * H_;
  const bf16_t* Wtv = Wt + (size_t)2 * E_ * H_;

  v8f accq[4], acck[4], accv[4];
#pragma unroll
  for (int t = 0; t < 4; ++t) { accq[t] = vzero8(); acck[t] = vzero8(); accv[t] = vzero8(); }

  const float* xrows = x + (size_t)ib * E_;
  v16bf a = load_x_afrag(xrows, 0, lane);  // prologue: first A-frag

  for (int e0 = 0; e0 < E_; e0 += 32) {
    // Issue next chunk's A-frag loads early (HBM latency hidden by 12 WMMAs).
    v16bf anext = a;
    if (e0 + 32 < E_) anext = load_x_afrag(xrows, e0 + 32, lane);

    // Preload first B-frags of this chunk.
    v16bf bq = load_wt_bfrag(Wtq, e0, 0, lane);
    v16bf bk = load_wt_bfrag(Wtk, e0, 0, lane);
    v16bf bv = load_wt_bfrag(Wtv, e0, 0, lane);

#pragma unroll
    for (int t = 0; t < 4; ++t) {
      v16bf nq = bq, nk = bk, nv = bv;
      if (t < 3) {  // issue tile t+1's loads before tile t's WMMAs
        nq = load_wt_bfrag(Wtq, e0, t + 1, lane);
        nk = load_wt_bfrag(Wtk, e0, t + 1, lane);
        nv = load_wt_bfrag(Wtv, e0, t + 1, lane);
      }
      accq[t] = WMMA_BF16(a, bq, accq[t]);
      acck[t] = WMMA_BF16(a, bk, acck[t]);
      accv[t] = WMMA_BF16(a, bv, accv[t]);
      bq = nq; bk = nk; bv = nv;
    }
    a = anext;
  }

  const float qscale = 0.125f;  // 1/sqrt(64)
#pragma unroll
  for (int t = 0; t < 4; ++t) {
#pragma unroll
    for (int r = 0; r < 8; ++r) {
      const int row = ib + r + 8 * hi;
      const int col = 16 * t + n;
      qh[(size_t)row * H_ + col] = (bf16_t)(accq[t][r] * qscale);
      kh[(size_t)row * H_ + col] = (bf16_t)(acck[t][r]);
      const int b  = row >> 12;          // row / T_
      const int rl = row & (T_ - 1);     // row % T_
      vth[((size_t)b * H_ + col) * T_ + rl] = (bf16_t)(accv[t][r]);
    }
  }
}

// ---------------------------------------------------------------------------
// Kernel 2: flash attention. One wave = 16 query rows, chunks of 32 keys.
// ---------------------------------------------------------------------------
__global__ __launch_bounds__(128) void attn_kernel(
    const bf16_t* __restrict__ qh, const bf16_t* __restrict__ kh,
    const bf16_t* __restrict__ vth, float* __restrict__ out) {
  __shared__ bf16_t ldsP[4][16 * 32];  // per-wave P tile (16 q x 32 keys)

  const int lane = threadIdx.x & 31;
  const int wv   = threadIdx.x >> 5;
  const int tile = blockIdx.x * 4 + wv;  // 0..1023
  const int b    = tile >> 8;            // 256 tiles per batch
  const int qb   = (tile & 255) * 16;    // query base within batch
  const int n    = lane & 15;
  const int hi   = lane >> 4;
  const int grow = b * T_ + qb;          // global base row

  // Q fragments (A-layout), H split 0..31 / 32..63.
  const bf16_t* qrows = qh + (size_t)grow * H_;
  v16bf aq0 = load_a_bf16(qrows, H_, lane);
  v16bf aq1 = load_a_bf16(qrows + 32, H_, lane);

  // Constant all-ones B-fragment: rowsum(P) = P @ ones via the matrix core.
  v16bf ones;
#pragma unroll
  for (int j = 0; j < 16; ++j) ones[j] = (bf16_t)1.0f;

  const bf16_t* kbase = kh + (size_t)b * T_ * H_;
  const bf16_t* vbase = vth + (size_t)b * H_ * T_;
  bf16_t* Pl = &ldsP[wv][0];

  v8f o[4];
  float mr[8], lr[8];
#pragma unroll
  for (int t = 0; t < 4; ++t) o[t] = vzero8();
#pragma unroll
  for (int r = 0; r < 8; ++r) { mr[r] = -3.0e38f; lr[r] = 0.f; }

  const int nch = (qb + 16 + 31) >> 5;  // key chunks (causal)
  for (int c = 0; c < nch; ++c) {
    const int kb = c * 32;

    if (c + 1 < nch) {  // prefetch next chunk's K rows and V columns
      __builtin_prefetch((const void*)(kbase + (size_t)(kb + 32 + lane) * H_), 0, 1);
      __builtin_prefetch((const void*)(vbase + (size_t)lane * T_ + kb + 32), 0, 1);
      __builtin_prefetch((const void*)(vbase + (size_t)(lane + 32) * T_ + kb + 32), 0, 1);
    }

    // S = Q @ K^T for two 16-key subtiles (B-frag: contiguous 32B along H).
    v8f s[2];
#pragma unroll
    for (int u = 0; u < 2; ++u) {
      const bf16_t* kr = kbase + (size_t)(kb + 16 * u + n) * H_ + 16 * hi;
      v16bf bk0 = *(const v16bf*)(kr);       // H 0..31
      v16bf bk1 = *(const v16bf*)(kr + 32);  // H 32..63
      v8f acc = vzero8();
      acc = WMMA_BF16(aq0, bk0, acc);
      acc = WMMA_BF16(aq1, bk1, acc);
      s[u] = acc;
    }

    // Causal mask (only the diagonal chunk needs it; kb <= qb always holds).
    if (kb + 31 > qb) {
#pragma unroll
      for (int u = 0; u < 2; ++u) {
        const int key = kb + 16 * u + n;
#pragma unroll
        for (int r = 0; r < 8; ++r) {
          const int qr = qb + r + 8 * hi;
          s[u][r] = (key > qr) ? -3.0e38f : s[u][r];
        }
      }
    }

    // Online softmax (max part). Row M = r + 8*hi lives across the 16 lanes
    // sharing hi, so xor-shuffles 1/2/4/8 reduce within the right lane group.
    float al[8];
#pragma unroll
    for (int r = 0; r < 8; ++r) {
      float rm = fmaxf(s[0][r], s[1][r]);
      rm = fmaxf(rm, __shfl_xor(rm, 1));
      rm = fmaxf(rm, __shfl_xor(rm, 2));
      rm = fmaxf(rm, __shfl_xor(rm, 4));
      rm = fmaxf(rm, __shfl_xor(rm, 8));
      const float mn = fmaxf(mr[r], rm);
      al[r] = __expf(mr[r] - mn);
      mr[r] = mn;
      s[0][r] = __expf(s[0][r] - mn);
      s[1][r] = __expf(s[1][r] - mn);
#pragma unroll
      for (int t = 0; t < 4; ++t) o[t][r] *= al[r];
    }

    // C-layout P -> row-major LDS (wave-private; DS ops are in-order in-wave).
#pragma unroll
    for (int r = 0; r < 8; ++r) {
      Pl[(r + 8 * hi) * 32 + n]      = (bf16_t)s[0][r];
      Pl[(r + 8 * hi) * 32 + 16 + n] = (bf16_t)s[1][r];
    }

    // P back in A-layout from LDS.
    v16bf ap = load_a_bf16(Pl, 32, lane);

    // Row sums via the matrix core: every column of P @ ones is rowsum(P),
    // landing already in C-layout (lane n irrelevant, vgpr r = row r+8*hi).
    v8f rsum = WMMA_BF16(ap, ones, vzero8());
#pragma unroll
    for (int r = 0; r < 8; ++r) lr[r] = lr[r] * al[r] + rsum[r];

    // P @ V (B-frag: contiguous 32B along T in vT).
#pragma unroll
    for (int t = 0; t < 4; ++t) {
      v16bf bv = *(const v16bf*)(vbase + (size_t)(16 * t + n) * T_ + kb + 16 * hi);
      o[t] = WMMA_BF16(ap, bv, o[t]);
    }
  }

  // Epilogue: normalize and store f32 output [B, T, H].
#pragma unroll
  for (int r = 0; r < 8; ++r) {
    const float inv = 1.0f / lr[r];
    const int row = grow + r + 8 * hi;
#pragma unroll
    for (int t = 0; t < 4; ++t) {
      out[(size_t)row * H_ + 16 * t + n] = o[t][r] * inv;
    }
  }
}

// ---------------------------------------------------------------------------
extern "C" void kernel_launch(void* const* d_in, const int* in_sizes, int n_in,
                              void* d_out, int out_size, void* d_ws, size_t ws_size,
                              hipStream_t stream) {
  const float* x  = (const float*)d_in[0];
  const float* Wq = (const float*)d_in[1];
  const float* Wk = (const float*)d_in[2];
  const float* Wv = (const float*)d_in[3];
  float* out = (float*)d_out;

  // Workspace layout (bf16): q [NROW,H], k [NROW,H], vT [B,H,T], Wt [3,H,E].
  bf16_t* qh  = (bf16_t*)d_ws;
  bf16_t* kh  = qh + (size_t)NROW * H_;
  bf16_t* vth = kh + (size_t)NROW * H_;
  bf16_t* Wt  = vth + (size_t)NROW * H_;   // 3 * 128 KB

  // Kernel 0: weight convert+transpose.
  {
    const dim3 blk(256);
    const dim3 grd((E_ * H_) / 256, 3);
    wprep_kernel<<<grd, blk, 0, stream>>>(Wq, Wk, Wv, Wt);
  }

  const dim3 blk(128);                 // 4 waves (wave32)
  const dim3 grd(NROW / 16 / 4);       // 256 blocks -> 1024 wave-tiles

  qkv_proj_kernel<<<grd, blk, 0, stream>>>(x, Wt, qh, kh, vth);
  attn_kernel<<<grd, blk, 0, stream>>>(qh, kh, vth, out);
}
